// FloquetRecurrentSolver_83047487636114
// MI455X (gfx1250) — matrix operand order repaired and checked by reading, not compiled, not run on hardware
//
#include <hip/hip_runtime.h>

typedef unsigned short u16_t;
typedef __attribute__((ext_vector_type(16))) __bf16 bf16x16;
typedef __attribute__((ext_vector_type(8)))  float   f32x8;
typedef __attribute__((ext_vector_type(4)))  unsigned int u32x4;

#define NND 43008
#define NED 131072
#define NBG 1024

__device__ __forceinline__ u16_t f2bf(float f) {
  union { float f; unsigned u; } v; v.f = f;
  unsigned u = v.u;
  unsigned r = u + 0x7FFFu + ((u >> 16) & 1u);   // round-to-nearest-even
  return (u16_t)(r >> 16);
}
__device__ __forceinline__ float bf2f(u16_t h) {
  union { unsigned u; float f; } v; v.u = ((unsigned)h) << 16;
  return v.f;
}

union AFrag { bf16x16 v; u32x4 q[2]; };

// bf16 WMMA GEMM: out[M,128] = act(A[M,K] @ W[K,128] + bias), K = KSTEPS*32 (zero-padded via LDS guard)
// GATHER=1: A[row, c] = nodeF[srcIdx[row], c] (c < LDNG) | epad[row, c-LDNG] (32-col zero-padded eattr)
//           KSTEPS==9 -> LDNG=256 (conv1), KSTEPS==5 -> LDNG=128 (convs 2-5); classification is static.
// GATHER=0: plain bf16 A with leading dim lda (K a multiple of 32).
// Epilogue: outB != nullptr -> bf16 store (ldo, coloff); else atomicAdd f32 scatter by dstIdx.
template <int GATHER, int KSTEPS>
__global__ void __launch_bounds__(128) wmma_gemm_k(
    const u16_t* __restrict__ A, long lda,
    const u16_t* __restrict__ nodeF,
    const int* __restrict__ srcIdx,
    const u16_t* __restrict__ epad,
    int kreal,
    const float* __restrict__ W, const float* __restrict__ bias,
    int relu,
    u16_t* __restrict__ outB, long ldo, long coloff,
    float* __restrict__ outAgg, const int* __restrict__ dstIdx)
{
  constexpr int LDNG = (KSTEPS == 9) ? 256 : 128;   // gather node-feature width
  extern __shared__ u16_t smem[];   // B fragments, swizzled: [((nb*KSTEPS+ks)*32 + lane)*16 + j]
  const int tid = threadIdx.x;

  // ---- W (f32, row-major KxN=128) -> LDS bf16 in B-fragment layout; coalesced reads, b32 stores ----
  constexpr int PAIRS = KSTEPS * 2048;      // (KSTEPS*32/2) k-pairs * 128 cols
  for (int t = tid; t < PAIRS; t += 128) {
    const int kp   = t >> 7;                // pair of consecutive k (same 16-block, same lane)
    const int k    = kp * 2;
    const int ncol = t & 127;
    const int nb   = ncol >> 4;
    const int n    = ncol & 15;
    const int ks   = k >> 5;
    const int kin  = k & 31;
    const int lane = n | ((kin & 8) << 1);
    const int j    = (kin & 7) | ((kin & 16) >> 1);   // even
    const float w0 = (k     < kreal) ? W[(size_t)k * 128 + ncol]       : 0.0f;
    const float w1 = (k + 1 < kreal) ? W[(size_t)(k + 1) * 128 + ncol] : 0.0f;
    const unsigned pk = (unsigned)f2bf(w0) | ((unsigned)f2bf(w1) << 16);
    *(unsigned*)&smem[(((size_t)nb * KSTEPS + ks) * 32 + lane) * 16 + j] = pk;
  }
  __syncthreads();

  const int wave = tid >> 5;
  const int lane = tid & 31;
  const int rsub = lane & 15;
  const int hi   = (lane >> 4) & 1;

  for (int mt = 0; mt < 4; ++mt) {
    const long g   = ((long)blockIdx.x * 4 + wave) * 4 + mt;
    const long m0  = g * 16;
    const long row = m0 + rsub;

    const u16_t* nbase;
    const u16_t* erow = nullptr;
    if (GATHER) {
      nbase = nodeF + (size_t)srcIdx[row] * LDNG;
      erow  = epad + (size_t)row * 32;
    } else {
      nbase = A + (size_t)row * lda;
    }

    f32x8 acc[8];
#pragma unroll
    for (int nb = 0; nb < 8; ++nb) {
      f32x8 z = {0.f, 0.f, 0.f, 0.f, 0.f, 0.f, 0.f, 0.f};
      acc[nb] = z;
    }

#pragma unroll
    for (int ks = 0; ks < KSTEPS; ++ks) {
      // A fragment: classification is compile-time under full unroll
      AFrag af;
      if (GATHER && ks * 32 >= LDNG) {
        const int c0 = ks * 32 - LDNG + (hi ? 8 : 0);
        af.q[0] = *(const u32x4*)(erow + c0);
        af.q[1] = *(const u32x4*)(erow + c0 + 16);
      } else {
        const int c0 = ks * 32 + (hi ? 8 : 0);
        af.q[0] = *(const u32x4*)(nbase + c0);
        af.q[1] = *(const u32x4*)(nbase + c0 + 16);
      }
      // preload all 8 B fragments, then burst 8 WMMAs (staggered waits)
      AFrag bf[8];
#pragma unroll
      for (int nb = 0; nb < 8; ++nb) {
        const u16_t* bp = smem + (((size_t)nb * KSTEPS + ks) * 32 + lane) * 16;
        bf[nb].q[0] = *(const u32x4*)(bp);
        bf[nb].q[1] = *(const u32x4*)(bp + 8);
      }
#pragma unroll
      for (int nb = 0; nb < 8; ++nb) {
        acc[nb] = __builtin_amdgcn_wmma_f32_16x16x32_bf16(
            false, af.v, false, bf[nb].v, (short)0, acc[nb], false, false);
      }
    }

    if (outB) {
      size_t obase[8];
#pragma unroll
      for (int r = 0; r < 8; ++r)
        obase[r] = (size_t)(m0 + r + (hi ? 8 : 0)) * ldo + coloff;
#pragma unroll
      for (int nb = 0; nb < 8; ++nb) {
        const int col  = nb * 16 + rsub;
        const float bvv = bias[col];
#pragma unroll
        for (int r = 0; r < 8; ++r) {
          float v = acc[nb][r] + bvv;
          if (relu) v = v > 0.0f ? v : 0.0f;
          outB[obase[r] + col] = f2bf(v);
        }
      }
    } else {
      int drow[8];
#pragma unroll
      for (int r = 0; r < 8; ++r)
        drow[r] = dstIdx[m0 + r + (hi ? 8 : 0)];
#pragma unroll
      for (int nb = 0; nb < 8; ++nb) {
        const int col  = nb * 16 + rsub;
        const float bvv = bias[col];
#pragma unroll
        for (int r = 0; r < 8; ++r) {
          float v = acc[nb][r] + bvv;
          __hip_atomic_fetch_add(&outAgg[(size_t)drow[r] * 128 + col], v,
                                 __ATOMIC_RELAXED, __HIP_MEMORY_SCOPE_AGENT);
        }
      }
    }
  }
}

// edge_attr (E,16) f32 -> (E,32) bf16, cols 16..31 zero
__global__ void epad_k(const float* __restrict__ in, u16_t* __restrict__ outp) {
  long t = (long)blockIdx.x * blockDim.x + threadIdx.x;
  if (t >= (long)NED * 32) return;
  long e = t >> 5;
  int c = (int)(t & 31);
  outp[t] = (c < 16) ? f2bf(in[e * 16 + c]) : (u16_t)0;
}

__global__ void zero_f32_k(float* __restrict__ p, long n) {
  long t = (long)blockIdx.x * blockDim.x + threadIdx.x;
  if (t < n) p[t] = 0.0f;
}

// encoder layer 1 (K=4) with on-the-fly node feature build
__global__ void enc1_k(const float* __restrict__ x, const float* __restrict__ W,
                       const float* __restrict__ b, u16_t* __restrict__ outp, int ridx) {
  int t = blockIdx.x * blockDim.x + threadIdx.x;
  if (t >= NND * 128) return;
  int m = t >> 7, n = t & 127;
  int g = m / 42, nig = m - g * 42;
  float offs = x[(size_t)(g * 42 + ridx) * 3 + 0];
  float x0 = x[(size_t)m * 3 + 0];
  float x1 = x[(size_t)m * 3 + 1];
  float x2 = (nig == ridx) ? 1.0f : x[(size_t)m * 3 + 2];
  float a = b[n] + offs * W[n] + x0 * W[128 + n] + x1 * W[256 + n] + x2 * W[384 + n];
  outp[t] = f2bf(a > 0.0f ? a : 0.0f);
}

// hcat[:, 128:256] = broadcast memory state
__global__ void xm_fill_k(const float* __restrict__ xm, u16_t* __restrict__ hcat) {
  int t = blockIdx.x * blockDim.x + threadIdx.x;
  if (t >= NND * 128) return;
  int m = t >> 7, f = t & 127;
  int g = m / 42, q = m & 1;
  hcat[(size_t)m * 256 + 128 + f] = f2bf(xm[(size_t)g * 256 + q * 128 + f]);
}

// An[m, :] = [hPrev[m, 0:ldh] (bf16) | agg[m, 0:128] (f32->bf16)]
__global__ void buildAn_k(const u16_t* __restrict__ hPrev, int ldh,
                          const float* __restrict__ agg, u16_t* __restrict__ An, long total) {
  long t = (long)blockIdx.x * blockDim.x + threadIdx.x;
  if (t >= total) return;
  int ldA = ldh + 128;
  long m = t / ldA;
  int c = (int)(t - m * ldA);
  An[t] = (c < ldh) ? hPrev[(size_t)m * ldh + c] : f2bf(agg[(size_t)m * 128 + (c - ldh)]);
}

// decoder layer 1: emb = [offs, h_root(128)] (129) -> 256, ReLU
__global__ void dec1_k(const float* __restrict__ x, const u16_t* __restrict__ h,
                       const float* __restrict__ W, const float* __restrict__ b,
                       float* __restrict__ outp, int ridx) {
  int t = blockIdx.x * blockDim.x + threadIdx.x;
  if (t >= NBG * 256) return;
  int g = t >> 8, n = t & 255;
  long root = (long)g * 42 + ridx;
  float acc = b[n] + x[root * 3] * W[n];
  const u16_t* hp = h + (size_t)root * 128;
  for (int k = 0; k < 128; ++k) acc += bf2f(hp[k]) * W[(size_t)(k + 1) * 256 + n];
  outp[t] = acc > 0.0f ? acc : 0.0f;
}

__global__ void dec2_k(const float* __restrict__ t1, const float* __restrict__ W,
                       const float* __restrict__ b, float* __restrict__ outp) {
  int t = blockIdx.x * blockDim.x + threadIdx.x;
  if (t >= NBG * 64) return;
  int g = t >> 6, n = t & 63;
  float acc = b[n];
  const float* tp = t1 + (size_t)g * 256;
  for (int k = 0; k < 256; ++k) acc += tp[k] * W[(size_t)k * 64 + n];
  outp[t] = acc > 0.0f ? acc : 0.0f;
}

// decoder layer 3 + write matrix (B,2) and broadcast end_embedding (B,2,128)
__global__ void dec3_k(const float* __restrict__ t2, const float* __restrict__ W,
                       const float* __restrict__ b, float* __restrict__ outp, int i) {
  int t = blockIdx.x * blockDim.x + threadIdx.x;
  if (t >= NBG) return;
  float acc = b[0];
  const float* tp = t2 + (size_t)t * 64;
  for (int k = 0; k < 64; ++k) acc += tp[k] * W[k];
  outp[(size_t)t * 2 + i] = acc;
  float* ee = outp + 2048 + ((size_t)t * 2 + i) * 128;
  for (int f = 0; f < 128; ++f) ee[f] = acc;
}

extern "C" void kernel_launch(void* const* d_in, const int* in_sizes, int n_in,
                              void* d_out, int out_size, void* d_ws, size_t ws_size,
                              hipStream_t stream) {
  const int N = NND, E = NED, B = NBG;
  const float* x      = (const float*)d_in[0];
  const float* xmemo  = (const float*)d_in[1];
  const int*   eidx   = (const int*)d_in[2];     // (2, E) int32
  const float* eattrF = (const float*)d_in[3];
  auto P = [&](int i) { return (const float*)d_in[i]; };
  const int* srcI = eidx;
  const int* dstI = eidx + E;
  float* out = (float*)d_out;

  // ---- workspace carve (256B aligned) ----
  char* base = (char*)d_ws;
  size_t off = 0;
  auto carve = [&](size_t bytes) -> char* {
    char* p = base + off;
    off = (off + bytes + 255) & ~(size_t)255;
    return p;
  };
  u16_t* epadB  = (u16_t*)carve((size_t)E * 32 * 2);    // zero-padded edge attrs
  u16_t* tN1    = (u16_t*)carve((size_t)N * 128 * 2);
  u16_t* hcat   = (u16_t*)carve((size_t)N * 256 * 2);   // [enc_h | xm]
  u16_t* hbuf   = (u16_t*)carve((size_t)N * 128 * 2);
  u16_t* te1    = (u16_t*)carve((size_t)E * 128 * 2);
  u16_t* te2    = (u16_t*)carve((size_t)E * 128 * 2);
  float* agg    = (float*)carve((size_t)N * 128 * 4);
  u16_t* tn1    = (u16_t*)carve((size_t)N * 128 * 2);
  u16_t* tn2    = (u16_t*)carve((size_t)N * 128 * 2);
  float* dT1    = (float*)carve((size_t)B * 256 * 4);
  float* dT2    = (float*)carve((size_t)B * 64 * 4);
  u16_t* An     = te1;   // alias: te1 dead before An is built, An dead before te1 rewritten

  epad_k<<<dim3((unsigned)(((long)E * 32 + 255) / 256)), dim3(256), 0, stream>>>(eattrF, epadB);

  auto gemm = [&](const u16_t* A, long lda,
                  const u16_t* nodeF, const int* sIdx,
                  long M, int kreal,
                  const float* W, const float* bias, int relu,
                  u16_t* outB, long ldo, long coloff,
                  float* outAgg, const int* dIdx) {
    const int ksteps = (kreal + 31) / 32;
    const size_t shmem = (size_t)ksteps * 8192;
    const dim3 grid((unsigned)(M / 256)), blk(128);
    if (sIdx) {
      if (ksteps == 9)
        wmma_gemm_k<1, 9><<<grid, blk, shmem, stream>>>(A, lda, nodeF, sIdx, epadB, kreal,
                                                        W, bias, relu, outB, ldo, coloff, outAgg, dIdx);
      else
        wmma_gemm_k<1, 5><<<grid, blk, shmem, stream>>>(A, lda, nodeF, sIdx, epadB, kreal,
                                                        W, bias, relu, outB, ldo, coloff, outAgg, dIdx);
    } else {
      if (ksteps == 4)
        wmma_gemm_k<0, 4><<<grid, blk, shmem, stream>>>(A, lda, nodeF, sIdx, epadB, kreal,
                                                        W, bias, relu, outB, ldo, coloff, outAgg, dIdx);
      else if (ksteps == 8)
        wmma_gemm_k<0, 8><<<grid, blk, shmem, stream>>>(A, lda, nodeF, sIdx, epadB, kreal,
                                                        W, bias, relu, outB, ldo, coloff, outAgg, dIdx);
      else
        wmma_gemm_k<0, 12><<<grid, blk, shmem, stream>>>(A, lda, nodeF, sIdx, epadB, kreal,
                                                         W, bias, relu, outB, ldo, coloff, outAgg, dIdx);
    }
  };

  for (int i = 0; i < 2; ++i) {
    const int ridx = 20 + i;   // bz*dimq + i

    enc1_k<<<dim3((N * 128 + 255) / 256), dim3(256), 0, stream>>>(x, P(4), P(5), tN1, ridx);
    // encoder layer 2 -> hcat[:, 0:128]
    gemm(tN1, 128, nullptr, nullptr, N, 128, P(6), P(7), 0, hcat, 256, 0, nullptr, nullptr);
    xm_fill_k<<<dim3((N * 128 + 255) / 256), dim3(256), 0, stream>>>(xmemo, hcat);

    for (int l = 0; l < 5; ++l) {
      const int cb = 8 + l * 12;
      const u16_t* nodeF = (l == 0) ? hcat : hbuf;
      const long   ldn   = (l == 0) ? 256 : 128;
      const int    kEdge = (int)ldn + 16;    // 272 or 144 (zero-padded to 288/160 by guards)
      const int    kNode = (int)ldn + 128;   // 384 or 256

      zero_f32_k<<<dim3((unsigned)(((long)N * 128 + 255) / 256)), dim3(256), 0, stream>>>(
          agg, (long)N * 128);

      // mlp1 on edges, gather fused, scatter-add fused into layer 3
      gemm(nullptr, 0, nodeF, srcI, E, kEdge, P(cb + 0), P(cb + 1), 1, te1, 128, 0, nullptr, nullptr);
      gemm(te1, 128, nullptr, nullptr, E, 128, P(cb + 2), P(cb + 3), 1, te2, 128, 0, nullptr, nullptr);
      gemm(te2, 128, nullptr, nullptr, E, 128, P(cb + 4), P(cb + 5), 0, nullptr, 0, 0, agg, dstI);

      // mlp2 on nodes
      buildAn_k<<<dim3((unsigned)(((long)N * kNode + 255) / 256)), dim3(256), 0, stream>>>(
          nodeF, (int)ldn, agg, An, (long)N * kNode);
      gemm(An, kNode, nullptr, nullptr, N, kNode, P(cb + 6), P(cb + 7), 1, tn1, 128, 0, nullptr, nullptr);
      gemm(tn1, 128, nullptr, nullptr, N, 128, P(cb + 8), P(cb + 9), 1, tn2, 128, 0, nullptr, nullptr);
      gemm(tn2, 128, nullptr, nullptr, N, 128, P(cb + 10), P(cb + 11), 0, hbuf, 128, 0, nullptr, nullptr);
    }

    dec1_k<<<dim3((B * 256 + 255) / 256), dim3(256), 0, stream>>>(x, hbuf, P(68), P(69), dT1, ridx);
    dec2_k<<<dim3((B * 64 + 255) / 256), dim3(256), 0, stream>>>(dT1, P(70), P(71), dT2);
    dec3_k<<<dim3((B + 255) / 256), dim3(256), 0, stream>>>(dT2, P(72), P(73), out, i);
  }

  (void)in_sizes; (void)n_in; (void)out_size; (void)ws_size;
}
